// UniTR_6425271075227
// MI455X (gfx1250) — compile-verified, matrix operand order checked by reading, not run on hardware
//
#include <hip/hip_runtime.h>
#include <math.h>

#define D_    128
#define H_    8
#define DH    16
#define DFF_  256
#define S_    1000
#define L_    90
#define N_    (S_*L_)
#define VN_   45000
#define LP    96          // padded set length (6 tiles of 16)

typedef _Float16 half_t;
typedef __attribute__((ext_vector_type(16))) _Float16 v16h;
typedef __attribute__((ext_vector_type(8)))  float    v8f;

// f16 weight layout in workspace (element offsets, halves)
#define WH_INPROJ 0
#define WH_OUT    49152
#define WH_L1     65536
#define WH_L2     98304
#define WH_LL1    131072
#define WH_LL2    163840
#define WH_TOTAL  196608
#define SRC2_BYTE_OFF ((size_t)WH_TOTAL * 2)   // 393216 B, 256-aligned

#define WMMA_F16(a,b,c) __builtin_amdgcn_wmma_f32_16x16x32_f16(false,(a),false,(b),(short)0,(c),false,false)

__device__ __forceinline__ int kmap(int j, int hi) {
    return ((j < 4) ? (2 * j) : (2 * j + 8)) + (hi << 3);
}

// A-style fragment: per-lane row = lane&15, K along fastest dim.
// Works for A (rows = M) and for B when B[k][n] = W[n][k] (rows = out features).
__device__ __forceinline__ v16h ld_rowK(const half_t* p, int ld, int kb, int lane, int kvalid) {
    const int r = lane & 15, hi = lane >> 4;
    v16h f;
#pragma unroll
    for (int j = 0; j < 8; ++j) {
        int k = kmap(j, hi);
        half_t a = (half_t)0, b = (half_t)0;
        if (k     < kvalid) a = p[r * ld + kb + k];
        if (k + 1 < kvalid) b = p[r * ld + kb + k + 1];
        f[2 * j]     = a;
        f[2 * j + 1] = b;
    }
    return f;
}

// B fragment from a row-major [K][N-cols] matrix: per-lane col = lane&15, K strided by ld.
__device__ __forceinline__ v16h ld_colK(const half_t* p, int ld, int kb, int lane) {
    const int c = lane & 15, hi = lane >> 4;
    v16h f;
#pragma unroll
    for (int j = 0; j < 8; ++j) {
        int k = kb + kmap(j, hi);
        f[2 * j]     = p[k * ld + c];
        f[2 * j + 1] = p[(k + 1) * ld + c];
    }
    return f;
}

// ---------------- weight prep: f32 -> f16, fold 0.25 scale into Q proj ----------------
__global__ void prep_kernel(const float* __restrict__ ipw, const float* __restrict__ ow,
                            const float* __restrict__ l1, const float* __restrict__ l2,
                            const float* __restrict__ ll1, const float* __restrict__ ll2,
                            half_t* __restrict__ wh) {
    int i = blockIdx.x * blockDim.x + threadIdx.x;
    if (i >= WH_TOTAL) return;
    float v;
    if (i < 49152)        v = ipw[i] * ((i < 16384) ? 0.25f : 1.0f);  // Q rows pre-scaled
    else if (i < 65536)   v = ow [i - 49152];
    else if (i < 98304)   v = l1 [i - 65536];
    else if (i < 131072)  v = l2 [i - 98304];
    else if (i < 163840)  v = ll1[i - 131072];
    else                  v = ll2[i - 163840];
    wh[i] = (half_t)v;
}

// ---------------- attention kernel: one block per set ----------------
__global__ __launch_bounds__(128)
void attn_kernel(const float* __restrict__ src, const float* __restrict__ pos,
                 const int* __restrict__ inds, const unsigned char* __restrict__ masks,
                 const float* __restrict__ ipb, const float* __restrict__ outb,
                 const half_t* __restrict__ wh, float* __restrict__ src2) {
    extern __shared__ char smem[];
    half_t* sQK = (half_t*)smem;            // [96][128] q/k input (src+pos)
    half_t* sV  = sQK + LP * D_;            // [96][128] v input (src)
    half_t* sQP = sV  + LP * D_;            // [96][128]
    half_t* sKP = sQP + LP * D_;
    half_t* sVP = sKP + LP * D_;
    half_t* sO  = sVP + LP * D_;            // [96][128] attention output
    float*  sS  = (float*)(sO + LP * D_);   // [96][96] scores f32
    half_t* sA  = (half_t*)(sS + LP * LP);  // [96][96] attn f16

    const int set  = blockIdx.x;
    const int tid  = threadIdx.x;
    const int lane = tid & 31;
    const int wv   = tid >> 5;

    __builtin_prefetch(wh, 0, 0);
    __builtin_prefetch(wh + WH_OUT, 0, 0);

    // phase 1: gather + convert (rows 90..95 zero-padded)
    for (int idx = tid; idx < LP * (D_ / 4); idx += 128) {
        int r = idx >> 5, c4 = idx & 31, c = c4 * 4;
        if (r < L_) {
            int t = inds[set * L_ + r];
            float4 s4 = ((const float4*)src)[t * 32 + c4];
            float4 p4 = ((const float4*)pos)[t * 32 + c4];
            sQK[r * D_ + c + 0] = (half_t)(s4.x + p4.x);
            sQK[r * D_ + c + 1] = (half_t)(s4.y + p4.y);
            sQK[r * D_ + c + 2] = (half_t)(s4.z + p4.z);
            sQK[r * D_ + c + 3] = (half_t)(s4.w + p4.w);
            sV [r * D_ + c + 0] = (half_t)s4.x;
            sV [r * D_ + c + 1] = (half_t)s4.y;
            sV [r * D_ + c + 2] = (half_t)s4.z;
            sV [r * D_ + c + 3] = (half_t)s4.w;
        } else {
#pragma unroll
            for (int u = 0; u < 4; ++u) {
                sQK[r * D_ + c + u] = (half_t)0;
                sV [r * D_ + c + u] = (half_t)0;
            }
        }
    }
    __syncthreads();

    // phase 2: QKV projection, 3 mats x 6 Mtiles x 8 Ntiles = 144 jobs
    for (int job = wv; job < 144; job += 4) {
        int mat = job / 48, rem = job % 48;
        int m = rem >> 3, n = rem & 7;
        const half_t* Ain = (mat < 2 ? sQK : sV) + m * 16 * D_;
        const half_t* Bw  = wh + WH_INPROJ + (mat * D_ + n * 16) * D_;
        v8f acc = {};
#pragma unroll
        for (int kb = 0; kb < D_; kb += 32) {
            v16h a = ld_rowK(Ain, D_, kb, lane, 32);
            v16h b = ld_rowK(Bw,  D_, kb, lane, 32);
            acc = WMMA_F16(a, b, acc);
        }
        int colg = n * 16 + (lane & 15);
        float bias = ipb[mat * D_ + colg];
        if (mat == 0) bias *= 0.25f;                     // Q bias scaled like Q weights
        half_t* dst = (mat == 0) ? sQP : ((mat == 1) ? sKP : sVP);
        int rbase = m * 16 + ((lane >> 4) << 3);
#pragma unroll
        for (int j = 0; j < 8; ++j)
            dst[(rbase + j) * D_ + colg] = (half_t)(acc[j] + bias);
    }
    __syncthreads();

    // phase 3: per-head attention
    for (int h = 0; h < H_; ++h) {
        const int hb = h * DH;
        // (a) scores = Q Kt (K=16 zero-padded to 32), + mask
        for (int job = wv; job < 36; job += 4) {
            int m = job / 6, n = job % 6;
            v16h a = ld_rowK(sQP + m * 16 * D_, D_, hb, lane, 16);
            v16h b = ld_rowK(sKP + n * 16 * D_, D_, hb, lane, 16);
            v8f acc = {};
            acc = WMMA_F16(a, b, acc);
            int key = n * 16 + (lane & 15);
            float madd = (key < L_) ? (masks[set * L_ + key] ? -1e9f : 0.0f) : -1e9f;
            int rbase = m * 16 + ((lane >> 4) << 3);
#pragma unroll
            for (int j = 0; j < 8; ++j)
                sS[(rbase + j) * LP + key] = acc[j] + madd;
        }
        __syncthreads();
        // (b) softmax, one row per thread
        if (tid < LP) {
            float mx = -3.4e38f;
            for (int k2 = 0; k2 < LP; ++k2) mx = fmaxf(mx, sS[tid * LP + k2]);
            float sum = 0.0f;
            for (int k2 = 0; k2 < LP; ++k2) {
                float e = __expf(sS[tid * LP + k2] - mx);
                sum += e;
                sA[tid * LP + k2] = (half_t)e;
            }
            float inv = 1.0f / sum;
            for (int k2 = 0; k2 < LP; ++k2)
                sA[tid * LP + k2] = (half_t)((float)sA[tid * LP + k2] * inv);
        }
        __syncthreads();
        // (c) o = attn @ Vh (K = 96)
        for (int job = wv; job < 6; job += 4) {
            int m = job;
            v8f acc = {};
#pragma unroll
            for (int kb = 0; kb < LP; kb += 32) {
                v16h a = ld_rowK(sA + m * 16 * LP, LP, kb, lane, 32);
                v16h b = ld_colK(sVP + hb, D_, kb, lane);
                acc = WMMA_F16(a, b, acc);
            }
            int rbase = m * 16 + ((lane >> 4) << 3);
            int col = hb + (lane & 15);
#pragma unroll
            for (int j = 0; j < 8; ++j)
                sO[(rbase + j) * D_ + col] = (half_t)acc[j];
        }
    }
    __syncthreads();

    // phase 4: out projection + scatter (inds is a permutation)
    for (int job = wv; job < 48; job += 4) {
        int m = job >> 3, n = job & 7;
        v8f acc = {};
#pragma unroll
        for (int kb = 0; kb < D_; kb += 32) {
            v16h a = ld_rowK(sO + m * 16 * D_, D_, kb, lane, 32);
            v16h b = ld_rowK(wh + WH_OUT + (n * 16) * D_, D_, kb, lane, 32);
            acc = WMMA_F16(a, b, acc);
        }
        int colg = n * 16 + (lane & 15);
        float bias = outb[colg];
        int rbase = m * 16 + ((lane >> 4) << 3);
#pragma unroll
        for (int j = 0; j < 8; ++j) {
            int r = rbase + j;
            if (r < L_) {
                int t = inds[set * L_ + r];
                src2[t * D_ + colg] = acc[j] + bias;
            }
        }
    }
}

// ---------------- FFN + norms kernel: 96 tokens per block ----------------
__global__ __launch_bounds__(128)
void ffn_kernel(const float* __restrict__ src, const float* __restrict__ src2,
                const half_t* __restrict__ wh,
                const float* __restrict__ l1b, const float* __restrict__ l2b,
                const float* __restrict__ ll1b, const float* __restrict__ ll2b,
                const float* __restrict__ n1g, const float* __restrict__ n1b,
                const float* __restrict__ n2g, const float* __restrict__ n2b,
                const float* __restrict__ ln1g, const float* __restrict__ ln1b,
                const float* __restrict__ ln2g, const float* __restrict__ ln2b,
                const float* __restrict__ eg, const float* __restrict__ eb,
                float* __restrict__ out) {
    extern __shared__ char smem[];
    float*  xf = (float*)smem;              // [96][128] f32
    half_t* xh = (half_t*)(xf + LP * D_);   // [96][128] f16 (normalized)
    half_t* hh = xh + LP * D_;              // [96][256] f16 hidden

    const int tid = threadIdx.x, lane = tid & 31, wv = tid >> 5;
    const int base = blockIdx.x * LP;

    // step 1: x = src + src2
    for (int idx = tid; idx < LP * (D_ / 4); idx += 128) {
        int r = idx >> 5, c4 = idx & 31;
        int t = base + r;
        float4 v = make_float4(0.f, 0.f, 0.f, 0.f);
        if (t < N_) {
            float4 a = ((const float4*)src )[t * 32 + c4];
            float4 b = ((const float4*)src2)[t * 32 + c4];
            v = make_float4(a.x + b.x, a.y + b.y, a.z + b.z, a.w + b.w);
        }
        ((float4*)xf)[idx] = v;
    }
    __syncthreads();

    // step 2: LN1 (branch by token index), keep f32 + f16 copies
    if (tid < LP) {
        int t = base + tid;
        if (t < N_) {
            bool lid = (t < VN_);
            const float* g = lid ? ln1g : n1g;
            const float* b = lid ? ln1b : n1b;
            float mean = 0.f;
            for (int c = 0; c < D_; ++c) mean += xf[tid * D_ + c];
            mean *= (1.0f / D_);
            float var = 0.f;
            for (int c = 0; c < D_; ++c) { float d = xf[tid * D_ + c] - mean; var += d * d; }
            var *= (1.0f / D_);
            float inv = rsqrtf(var + 1e-5f);
            for (int c = 0; c < D_; ++c) {
                float y = (xf[tid * D_ + c] - mean) * inv * g[c] + b[c];
                xf[tid * D_ + c] = y;
                xh[tid * D_ + c] = (half_t)y;
            }
        } else {
            for (int c = 0; c < D_; ++c) { xf[tid * D_ + c] = 0.f; xh[tid * D_ + c] = (half_t)0; }
        }
    }
    __syncthreads();

    // step 3: h = relu(xn @ W1^T + b1), 6 x 16 = 96 jobs; per-tile branch, mixed tile does both
    for (int job = wv; job < 96; job += 4) {
        int m = job >> 4, n = job & 15;
        int t0 = base + m * 16;
        bool needL = (t0 < VN_), needI = !(t0 + 15 < VN_);
        v8f accL = {}, accI = {};
#pragma unroll
        for (int kb = 0; kb < D_; kb += 32) {
            v16h a = ld_rowK(xh + m * 16 * D_, D_, kb, lane, 32);
            if (needL) { v16h b = ld_rowK(wh + WH_LL1 + n * 16 * D_, D_, kb, lane, 32); accL = WMMA_F16(a, b, accL); }
            if (needI) { v16h b = ld_rowK(wh + WH_L1  + n * 16 * D_, D_, kb, lane, 32); accI = WMMA_F16(a, b, accI); }
        }
        int colg = n * 16 + (lane & 15);
        float bL = ll1b[colg], bI = l1b[colg];
        int rbase = m * 16 + ((lane >> 4) << 3);
#pragma unroll
        for (int j = 0; j < 8; ++j) {
            int t = base + rbase + j;
            float v = (t < VN_) ? (accL[j] + bL) : (accI[j] + bI);
            hh[(rbase + j) * DFF_ + colg] = (half_t)fmaxf(v, 0.f);
        }
    }
    __syncthreads();

    // step 4: f = h @ W2^T + b2; x2 = xn + f (in-place on xf)
    for (int job = wv; job < 48; job += 4) {
        int m = job >> 3, n = job & 7;
        int t0 = base + m * 16;
        bool needL = (t0 < VN_), needI = !(t0 + 15 < VN_);
        v8f accL = {}, accI = {};
#pragma unroll
        for (int kb = 0; kb < DFF_; kb += 32) {
            v16h a = ld_rowK(hh + m * 16 * DFF_, DFF_, kb, lane, 32);
            if (needL) { v16h b = ld_rowK(wh + WH_LL2 + n * 16 * DFF_, DFF_, kb, lane, 32); accL = WMMA_F16(a, b, accL); }
            if (needI) { v16h b = ld_rowK(wh + WH_L2  + n * 16 * DFF_, DFF_, kb, lane, 32); accI = WMMA_F16(a, b, accI); }
        }
        int colg = n * 16 + (lane & 15);
        float bL = ll2b[colg], bI = l2b[colg];
        int rbase = m * 16 + ((lane >> 4) << 3);
#pragma unroll
        for (int j = 0; j < 8; ++j) {
            int r = rbase + j;
            int t = base + r;
            float v = (t < VN_) ? (accL[j] + bL) : (accI[j] + bI);
            xf[r * D_ + colg] += v;
        }
    }
    __syncthreads();

    // step 5: LN2, + identity, final encoder LN
    if (tid < LP) {
        int t = base + tid;
        if (t < N_) {
            bool lid = (t < VN_);
            const float* g2 = lid ? ln2g : n2g;
            const float* b2 = lid ? ln2b : n2b;
            float mean = 0.f;
            for (int c = 0; c < D_; ++c) mean += xf[tid * D_ + c];
            mean *= (1.0f / D_);
            float var = 0.f;
            for (int c = 0; c < D_; ++c) { float d = xf[tid * D_ + c] - mean; var += d * d; }
            var *= (1.0f / D_);
            float inv = rsqrtf(var + 1e-5f);
            float m2 = 0.f, s2 = 0.f;
            for (int c = 0; c < D_; ++c) {
                float z = (xf[tid * D_ + c] - mean) * inv * g2[c] + b2[c] + src[t * D_ + c];
                xf[tid * D_ + c] = z;
                m2 += z; s2 += z * z;
            }
            m2 *= (1.0f / D_);
            float v2 = s2 * (1.0f / D_) - m2 * m2;
            float inv2 = rsqrtf(v2 + 1e-5f);
            for (int c = 0; c < D_; ++c)
                out[t * D_ + c] = (xf[tid * D_ + c] - m2) * inv2 * eg[c] + eb[c];
        }
    }
}

extern "C" void kernel_launch(void* const* d_in, const int* in_sizes, int n_in,
                              void* d_out, int out_size, void* d_ws, size_t ws_size,
                              hipStream_t stream) {
    (void)in_sizes; (void)n_in; (void)out_size; (void)ws_size;
    const float* src  = (const float*)d_in[0];
    const float* pos  = (const float*)d_in[1];
    const int*   inds = (const int*)d_in[2];
    const unsigned char* masks = (const unsigned char*)d_in[3];
    // d_in[4] = voxel_num (fixed 45000, hardcoded)
    const float* ipw  = (const float*)d_in[5];
    const float* ipb  = (const float*)d_in[6];
    const float* ow   = (const float*)d_in[7];
    const float* ob   = (const float*)d_in[8];
    const float* l1w  = (const float*)d_in[9];
    const float* l1b  = (const float*)d_in[10];
    const float* l2w  = (const float*)d_in[11];
    const float* l2b  = (const float*)d_in[12];
    const float* n1g  = (const float*)d_in[13];
    const float* n1b  = (const float*)d_in[14];
    const float* n2g  = (const float*)d_in[15];
    const float* n2b  = (const float*)d_in[16];
    const float* ll1w = (const float*)d_in[17];
    const float* ll1b = (const float*)d_in[18];
    const float* ll2w = (const float*)d_in[19];
    const float* ll2b = (const float*)d_in[20];
    const float* ln1g = (const float*)d_in[21];
    const float* ln1b = (const float*)d_in[22];
    const float* ln2g = (const float*)d_in[23];
    const float* ln2b = (const float*)d_in[24];
    const float* eg   = (const float*)d_in[25];
    const float* eb   = (const float*)d_in[26];

    half_t* wh   = (half_t*)d_ws;
    float*  src2 = (float*)((char*)d_ws + SRC2_BYTE_OFF);
    float*  out  = (float*)d_out;

    prep_kernel<<<(WH_TOTAL + 255) / 256, 256, 0, stream>>>(ipw, ow, l1w, l2w, ll1w, ll2w, wh);

    const int smemA = 6 * LP * D_ * 2 + LP * LP * 4 + LP * LP * 2;   // 202752 B
    hipFuncSetAttribute((const void*)attn_kernel, hipFuncAttributeMaxDynamicSharedMemorySize, smemA);
    attn_kernel<<<S_, 128, smemA, stream>>>(src, pos, inds, masks, ipb, ob, wh, src2);

    const int smemB = LP * D_ * 4 + LP * D_ * 2 + LP * DFF_ * 2;     // 122880 B
    hipFuncSetAttribute((const void*)ffn_kernel, hipFuncAttributeMaxDynamicSharedMemorySize, smemB);
    ffn_kernel<<<(N_ + LP - 1) / LP, 128, smemB, stream>>>(
        src, src2, wh, l1b, l2b, ll1b, ll2b,
        n1g, n1b, n2g, n2b, ln1g, ln1b, ln2g, ln2b, eg, eb, out);
}
